// KoopmanNet_14654428414590
// MI455X (gfx1250) — matrix operand order repaired and checked by reading, not compiled, not run on hardware
//
#include <hip/hip_runtime.h>
#include <hip/hip_bf16.h>

// ---------------------------------------------------------------------------
// Problem constants
// ---------------------------------------------------------------------------
static constexpr int BATCH = 16384;
static constexpr int TSTEP = 17;              // 1 + S
static constexpr int NDIM  = 128;
static constexpr int DDIM  = 5;
static constexpr int ROWS  = BATCH * TSTEP;   // 278528, divisible by 128

// ---------------------------------------------------------------------------
// Vector types
// ---------------------------------------------------------------------------
typedef __attribute__((ext_vector_type(16))) __bf16 v16bf;
typedef __attribute__((ext_vector_type(8)))  __bf16 bf8v;
typedef __attribute__((ext_vector_type(8)))  float  v8f;
typedef __attribute__((ext_vector_type(4)))  float  f4;
typedef bf8v bf8v_a __attribute__((may_alias));
typedef f4   f4_a   __attribute__((may_alias));

#define V8F_ZERO (v8f){0.f,0.f,0.f,0.f,0.f,0.f,0.f,0.f}

// Load a 16x32 bf16 A-fragment (weights, Wt[out][in] row-major, stride K).
// A layout used here (contiguous k, not AMD's table order): lane m=L&15, kh=L>>4; element e <-> K=kh*16+e.
// `wr` already includes (mt*16+j)*K + kh*16.
__device__ __forceinline__ v16bf load_a_frag(const __bf16* __restrict__ wr) {
  bf8v a0 = *(const bf8v_a*)wr;
  bf8v a1 = *(const bf8v_a*)(wr + 8);
  return __builtin_shufflevector(a0, a1, 0,1,2,3,4,5,6,7,8,9,10,11,12,13,14,15);
}

__device__ __forceinline__ v8f wmma_bf16(v16bf a, v16bf b, v8f c) {
  return __builtin_amdgcn_wmma_f32_16x16x32_bf16(false, a, false, b, (short)0, c,
                                                 false, false);
}

__device__ __forceinline__ void bias_relu(const float* __restrict__ bptr, int cbase,
                                          const v8f& acc, float (&out)[8]) {
  f4 b0 = *(const f4_a*)(bptr + cbase);
  f4 b1 = *(const f4_a*)(bptr + cbase + 4);
#pragma unroll
  for (int e = 0; e < 4; ++e) {
    out[e]     = fmaxf(acc[e]     + b0[e], 0.f);
    out[e + 4] = fmaxf(acc[e + 4] + b1[e], 0.f);
  }
}

// ---------------------------------------------------------------------------
// One hidden MLP layer, D = Wt_tile x act, with:
//  * rotating A-fragment prefetch: frags[kt] is consumed for tile mt while the
//    load for tile mt+1 is issued -> ~KT loads in flight, overlapping the WMMA
//    chain instead of load->wait(0)->wmma lockstep.
//  * fused transition: as soon as a channel-tile pair (2t, 2t+1) of the output
//    is finished (bias+relu), it is turned into the next layer's B-fragment
//    via two half-wave shuffles per element pair (no LDS, full EXEC).
// ---------------------------------------------------------------------------
template <int KT, int OT, int STRIDE>
__device__ __forceinline__ void wmma_layer_frags(
    const __bf16* __restrict__ Wt, int j, int kh,
    const v16bf (&bin)[KT], const float* __restrict__ bias,
    v16bf (&bout)[OT]) {
  const __bf16* wbase = Wt + (size_t)j * STRIDE + kh * 16;
  v16bf frags[KT];
#pragma unroll
  for (int kt = 0; kt < KT; ++kt)
    frags[kt] = load_a_frag(wbase + kt * 32);
  float vpair[2][8];
#pragma unroll
  for (int mt = 0; mt < 2 * OT; ++mt) {
    v8f acc = V8F_ZERO;
#pragma unroll
    for (int kt = 0; kt < KT; ++kt) {
      v16bf a = frags[kt];
      if (mt + 1 < 2 * OT)
        frags[kt] = load_a_frag(wbase + (mt + 1) * 16 * STRIDE + kt * 32);
      acc = wmma_bf16(a, bin[kt], acc);
    }
    bias_relu(bias, mt * 16 + kh * 8, acc, vpair[mt & 1]);
    if (mt & 1) {
      v16bf f;
#pragma unroll
      for (int e = 0; e < 8; ++e) {
        float v0 = vpair[0][e];
        float v1 = vpair[1][e];
        float s0 = __shfl_xor(v0, 16, 32);
        float s1 = __shfl_xor(v1, 16, 32);
        f[e]     = (__bf16)(kh ? s1 : v0);
        f[e + 8] = (__bf16)(kh ? v1 : s0);
      }
      bout[mt >> 1] = f;
    }
  }
}

// ---------------------------------------------------------------------------
// Weight prep: dst[n*Kpad + k] = (bf16) src[k*Nsrc + n], zero-padded.
// ---------------------------------------------------------------------------
__global__ void koop_prep(const float* __restrict__ src, __bf16* __restrict__ dst,
                          int Ksrc, int Nsrc, int Kpad, int Npad) {
  int o = blockIdx.x * blockDim.x + threadIdx.x;
  if (o >= Kpad * Npad) return;
  int k = o % Kpad;
  int n = o / Kpad;
  float v = (k < Ksrc && n < Nsrc) ? src[(size_t)k * Nsrc + n] : 0.f;
  dst[o] = (__bf16)v;
}

// ---------------------------------------------------------------------------
// Encoder: rows r = b*17 + t over x (B,T,N).  128 ->256 ->256 ->5.
// One wave per 16 rows; 8 waves / block; grid*8*16 == ROWS exactly.
// ---------------------------------------------------------------------------
__global__ __launch_bounds__(256) void koop_encoder(
    const float* __restrict__ x,
    const __bf16* __restrict__ W1t, const float* __restrict__ b1,
    const __bf16* __restrict__ W2t, const float* __restrict__ b2,
    const __bf16* __restrict__ W3t, const float* __restrict__ b3,
    float* __restrict__ g_out, float* __restrict__ lat) {
  const int lane = threadIdx.x & 31;
  const int gw   = blockIdx.x * 8 + (threadIdx.x >> 5);
  const int j    = lane & 15;
  const int kh   = lane >> 4;
  const int rowbase = gw * 16;

  // B-fragments straight from x (row j): element e <-> K = kh*16 + e
  v16bf bx[4];
  {
    const float* xr = x + (size_t)(rowbase + j) * NDIM + kh * 16;
#pragma unroll
    for (int kt = 0; kt < 4; ++kt) {
      v16bf f;
#pragma unroll
      for (int q = 0; q < 4; ++q) {
        f4 c = *(const f4_a*)(xr + kt * 32 + q * 4);
#pragma unroll
        for (int e = 0; e < 4; ++e) f[q * 4 + e] = (__bf16)c[e];
      }
      bx[kt] = f;
    }
  }

  v16bf bh1[8];
  wmma_layer_frags<4, 8, 128>(W1t, j, kh, bx, b1, bh1);   // 128 -> 256
  v16bf bh2[8];
  wmma_layer_frags<8, 8, 256>(W2t, j, kh, bh1, b2, bh2);  // 256 -> 256

  // ---- layer 3: 256 -> 5 via zero-padded 16x256 W3^T (one m-tile) ----
  v8f acc = V8F_ZERO;
  {
    const __bf16* wbase = W3t + (size_t)j * 256 + kh * 16;
#pragma unroll
    for (int kt = 0; kt < 8; ++kt)
      acc = wmma_bf16(load_a_frag(wbase + kt * 32), bh2[kt], acc);
  }
  // lane L holds batch-row j, channels d = e + 8*kh -> kh==0 lanes own d=0..4
  if (kh == 0) {
    int row = rowbase + j;            // row = b*17 + t
    int bb  = row / 17;
    int t   = row - bb * 17;
    float* gp = g_out + ((size_t)t * BATCH + bb) * 5;
    float o[5];
#pragma unroll
    for (int d = 0; d < 5; ++d) { o[d] = acc[d] + b3[d]; gp[d] = o[d]; }
    if (t == 0) {
      float* lp = lat + (size_t)bb * 5;
#pragma unroll
      for (int d = 0; d < 5; ++d) lp[d] = o[d];
    }
  }
}

// ---------------------------------------------------------------------------
// Scan: sequential 16-step latent dynamics, one thread per batch element.
// Weight reads are wave-uniform -> scalar K$ loads; all state in registers.
// ---------------------------------------------------------------------------
__device__ __forceinline__ void tiny_mlp_h2(float in,
    const float* __restrict__ W1, const float* __restrict__ B1,
    const float* __restrict__ W2, const float* __restrict__ B2,
    float (&h2)[32]) {
#pragma unroll
  for (int o = 0; o < 32; ++o) h2[o] = B2[o];
  for (int i = 0; i < 32; ++i) {
    float h1 = fmaxf(fmaf(in, W1[i], B1[i]), 0.f);
    const float* wr = W2 + i * 32;
#pragma unroll
    for (int o = 0; o < 32; ++o) h2[o] = fmaf(h1, wr[o], h2[o]);
  }
#pragma unroll
  for (int o = 0; o < 32; ++o) h2[o] = fmaxf(h2[o], 0.f);
}

__global__ __launch_bounds__(256) void koop_scan(
    const float* __restrict__ g0,
    const float* __restrict__ ocW1, const float* __restrict__ ocb1,
    const float* __restrict__ ocW2, const float* __restrict__ ocb2,
    const float* __restrict__ ocW3, const float* __restrict__ ocb3,
    const float* __restrict__ orW1, const float* __restrict__ orb1,
    const float* __restrict__ orW2, const float* __restrict__ orb2,
    const float* __restrict__ orW3, const float* __restrict__ orb3,
    float* __restrict__ lat) {
  int b = blockIdx.x * blockDim.x + threadIdx.x;
  const float* yp = g0 + (size_t)b * 5;
  float y0 = yp[0], y1 = yp[1], y2 = yp[2], y3 = yp[3], y4 = yp[4];
#pragma unroll 1
  for (int s = 1; s <= 16; ++s) {
    float n0, n1, n2, n3;
#pragma unroll
    for (int k = 0; k < 2; ++k) {
      float yr = k ? y2 : y0;
      float yi = k ? y3 : y1;
      float r2 = yr * yr + yi * yi;
      float h2[32];
      tiny_mlp_h2(r2, ocW1 + k * 32, ocb1 + k * 32, ocW2 + k * 1024,
                  ocb2 + k * 32, h2);
      const float* W3 = ocW3 + k * 64;
      float o0 = ocb3[k * 2], o1 = ocb3[k * 2 + 1];
#pragma unroll
      for (int i = 0; i < 32; ++i) {
        o0 = fmaf(h2[i], W3[2 * i], o0);
        o1 = fmaf(h2[i], W3[2 * i + 1], o1);
      }
      float sc  = __expf(fminf(fmaxf(o1, -5.f), 5.f) * 0.01f);
      float ang = o0 * 0.01f;
      float cc = sc * __cosf(ang), ss = sc * __sinf(ang);
      if (k == 0) { n0 = yr * cc - yi * ss; n1 = yr * ss + yi * cc; }
      else        { n2 = yr * cc - yi * ss; n3 = yr * ss + yi * cc; }
    }
    {
      float h2[32];
      tiny_mlp_h2(y4, orW1, orb1, orW2, orb2, h2);
      float o0 = orb3[0];
#pragma unroll
      for (int i = 0; i < 32; ++i) o0 = fmaf(h2[i], orW3[i], o0);
      y4 = __expf(fminf(fmaxf(o0, -5.f), 5.f) * 0.01f) * y4;
    }
    y0 = n0; y1 = n1; y2 = n2; y3 = n3;
    float* lp = lat + ((size_t)s * BATCH + b) * 5;
    lp[0] = y0; lp[1] = y1; lp[2] = y2; lp[3] = y3; lp[4] = y4;
  }
}

// ---------------------------------------------------------------------------
// Decoder: rows r = t*B + b over latents (17,B,5).  5 ->256 ->256 ->128.
// Layer 1 uses a zero-padded 256x32 W1^T so even the K=5 layer is WMMA.
// ---------------------------------------------------------------------------
__global__ __launch_bounds__(256) void koop_decoder(
    const float* __restrict__ lat,
    const __bf16* __restrict__ W1t, const float* __restrict__ b1,
    const __bf16* __restrict__ W2t, const float* __restrict__ b2,
    const __bf16* __restrict__ W3t, const float* __restrict__ b3,
    float* __restrict__ yp) {
  const int lane = threadIdx.x & 31;
  const int gw   = blockIdx.x * 8 + (threadIdx.x >> 5);
  const int j    = lane & 15;
  const int kh   = lane >> 4;
  const int rowbase = gw * 16;

  // Zero-padded latent B-fragment: element e <-> K = kh*16 + e; K<5 valid.
  v16bf bz[1];
  {
    const float* lr = lat + (size_t)(rowbase + j) * 5;
    float z[5];
#pragma unroll
    for (int d = 0; d < 5; ++d) z[d] = lr[d];
    v16bf f;
#pragma unroll
    for (int e = 0; e < 16; ++e) f[e] = (__bf16)0.f;
#pragma unroll
    for (int e = 0; e < 5; ++e) f[e] = (kh == 0) ? (__bf16)z[e] : (__bf16)0.f;
    bz[0] = f;
  }

  v16bf bh1[8];
  wmma_layer_frags<1, 8, 32>(W1t, j, kh, bz, b1, bh1);    // 5(pad32) -> 256
  v16bf bh2[8];
  wmma_layer_frags<8, 8, 256>(W2t, j, kh, bh1, b2, bh2);  // 256 -> 256

  // ---- layer 3: 256 -> 128, rotating prefetch, store straight to d_out ----
  float* outr = yp + (size_t)(rowbase + j) * NDIM;
  const __bf16* wbase = W3t + (size_t)j * 256 + kh * 16;
  v16bf frags[8];
#pragma unroll
  for (int kt = 0; kt < 8; ++kt) frags[kt] = load_a_frag(wbase + kt * 32);
#pragma unroll
  for (int mt = 0; mt < 8; ++mt) {
    v8f acc = V8F_ZERO;
#pragma unroll
    for (int kt = 0; kt < 8; ++kt) {
      v16bf a = frags[kt];
      if (mt + 1 < 8)
        frags[kt] = load_a_frag(wbase + (mt + 1) * 16 * 256 + kt * 32);
      acc = wmma_bf16(a, bh2[kt], acc);
    }
    int cb = mt * 16 + kh * 8;
    f4 bb0 = *(const f4_a*)(b3 + cb);
    f4 bb1 = *(const f4_a*)(b3 + cb + 4);
    f4 o0, o1;
#pragma unroll
    for (int e = 0; e < 4; ++e) {
      o0[e] = acc[e]     + bb0[e];
      o1[e] = acc[e + 4] + bb1[e];
    }
    *(f4_a*)(outr + cb)     = o0;
    *(f4_a*)(outr + cb + 4) = o1;
  }
}

// ---------------------------------------------------------------------------
// Launcher
// ---------------------------------------------------------------------------
extern "C" void kernel_launch(void* const* d_in, const int* in_sizes, int n_in,
                              void* d_out, int out_size, void* d_ws, size_t ws_size,
                              hipStream_t stream) {
  (void)in_sizes; (void)n_in; (void)out_size; (void)ws_size;

  const float* x      = (const float*)d_in[0];
  const float* enc_W1 = (const float*)d_in[1];
  const float* enc_b1 = (const float*)d_in[2];
  const float* enc_W2 = (const float*)d_in[3];
  const float* enc_b2 = (const float*)d_in[4];
  const float* enc_W3 = (const float*)d_in[5];
  const float* enc_b3 = (const float*)d_in[6];
  const float* dec_W1 = (const float*)d_in[7];
  const float* dec_b1 = (const float*)d_in[8];
  const float* dec_W2 = (const float*)d_in[9];
  const float* dec_b2 = (const float*)d_in[10];
  const float* dec_W3 = (const float*)d_in[11];
  const float* dec_b3 = (const float*)d_in[12];
  const float* oc_W1  = (const float*)d_in[13];
  const float* oc_b1  = (const float*)d_in[14];
  const float* oc_W2  = (const float*)d_in[15];
  const float* oc_b2  = (const float*)d_in[16];
  const float* oc_W3  = (const float*)d_in[17];
  const float* oc_b3  = (const float*)d_in[18];
  const float* or_W1  = (const float*)d_in[19];
  const float* or_b1  = (const float*)d_in[20];
  const float* or_W2  = (const float*)d_in[21];
  const float* or_b2  = (const float*)d_in[22];
  const float* or_W3  = (const float*)d_in[23];
  const float* or_b3  = (const float*)d_in[24];

  float* y_preds = (float*)d_out;                                   // (17,B,128)
  float* g_list  = y_preds + (size_t)TSTEP * BATCH * NDIM;          // (17,B,5)

  // Workspace layout
  char*  wsb = (char*)d_ws;
  float* lat = (float*)wsb;                                         // (17,B,5) f32
  size_t latBytes = (size_t)TSTEP * BATCH * DDIM * sizeof(float);   // 16B aligned
  __bf16* W1tE = (__bf16*)(wsb + latBytes);                         // (256,128)
  __bf16* W2tE = W1tE + 256 * 128;                                  // (256,256)
  __bf16* W2tD = W2tE + 256 * 256;                                  // (256,256)
  __bf16* W3tD = W2tD + 256 * 256;                                  // (128,256)
  __bf16* W3tE = W3tD + 128 * 256;                                  // (16,256) pad
  __bf16* W1tD = W3tE + 16 * 256;                                   // (256,32) pad

  // 1) weight prep (transpose + bf16 + zero-pad)
  koop_prep<<<(128 * 256 + 255) / 256, 256, 0, stream>>>(enc_W1, W1tE, 128, 256, 128, 256);
  koop_prep<<<(256 * 256 + 255) / 256, 256, 0, stream>>>(enc_W2, W2tE, 256, 256, 256, 256);
  koop_prep<<<(256 * 256 + 255) / 256, 256, 0, stream>>>(dec_W2, W2tD, 256, 256, 256, 256);
  koop_prep<<<(256 * 128 + 255) / 256, 256, 0, stream>>>(dec_W3, W3tD, 256, 128, 256, 128);
  koop_prep<<<(256 * 16  + 255) / 256, 256, 0, stream>>>(enc_W3, W3tE, 256, 5,   256, 16);
  koop_prep<<<(32 * 256  + 255) / 256, 256, 0, stream>>>(dec_W1, W1tD, 5,   256, 32,  256);

  // 2) encoder: ROWS/16 waves, 8 waves per block -> 2176 blocks exactly
  koop_encoder<<<ROWS / (16 * 8), 256, 0, stream>>>(
      x, W1tE, enc_b1, W2tE, enc_b2, W3tE, enc_b3, g_list, lat);

  // 3) sequential scan (reads g_list[t=0] slice, writes lat[1..16])
  koop_scan<<<BATCH / 256, 256, 0, stream>>>(
      g_list, oc_W1, oc_b1, oc_W2, oc_b2, oc_W3, oc_b3,
      or_W1, or_b1, or_W2, or_b2, or_W3, or_b3, lat);

  // 4) decoder
  koop_decoder<<<ROWS / (16 * 8), 256, 0, stream>>>(
      lat, W1tD, dec_b1, W2tD, dec_b2, W3tD, dec_b3, y_preds);
}
